// VqVae_11081015623975
// MI455X (gfx1250) — matrix operand order repaired and checked by reading, not compiled, not run on hardware
//
#include <hip/hip_runtime.h>

// ---------------------------------------------------------------------------
// Types for CDNA5 WMMA (wave32, v_wmma_f32_16x16x32_bf16)
// ---------------------------------------------------------------------------
typedef __bf16 bf16_t;
typedef bf16_t v16bf __attribute__((ext_vector_type(16)));
typedef bf16_t v8bf  __attribute__((ext_vector_type(8)));
typedef float  v8f   __attribute__((ext_vector_type(8)));

union FragBF { v16bf v; v8bf h[2]; };

#define BM 256
#define BN 128
#define BK 64
#define LDSP 72   // BK + 8 bf16 pad -> 144B row stride, kills LDS bank conflicts
#define MAXT 16

// ---------------------------------------------------------------------------
// Universal bf16 WMMA GEMM with fused gather-im2col + epilogue.
//   A[m][k] = act[ tabA[m*T + (k>>cgshift)] + (k & ((1<<cgshift)-1)) ] (0 if off<0)
//   B is pre-transposed weights: Bt[n][k]  (K-contiguous)
//   out = alpha*acc + bias[n] + skip[m*N+n]; f32 and/or bf16(+relu) stores,
//   row scatter via rowdst[m] (base element offset of output row).
// Requires: M % 256 == 0, N % 128 == 0, K % 64 == 0, CG in {64,256} so a BK
// window never crosses a tap boundary (single table entry per thread per step).
// Double-buffered LDS, one barrier per K-step, branchless masked A gathers.
// ---------------------------------------------------------------------------
__global__ __launch_bounds__(256) void gemm_bf16(
    const bf16_t* __restrict__ act, const int* __restrict__ tabA, int T, int cgshift,
    const bf16_t* __restrict__ Bt, int M, int N, int K,
    const float* __restrict__ bias, const float* __restrict__ skip,
    const int* __restrict__ rowdst, float alpha,
    float* __restrict__ outF, bf16_t* __restrict__ outB, int reluB)
{
  __shared__ bf16_t As[2][BM][LDSP];
  __shared__ bf16_t Bs[2][BN][LDSP];
  __shared__ int    Atab[BM][MAXT];

  const int tid  = threadIdx.x;
  const int lane = tid & 31;
  const int wave = tid >> 5;
  const int wm   = (wave & 3) * 64;   // 4 waves tile M (64 rows each)
  const int wn   = (wave >> 2) * 64;  // 2 waves tile N (64 cols each)
  const int m0   = blockIdx.y * BM;
  const int n0   = blockIdx.x * BN;
  const int cgmask = (1 << cgshift) - 1;
  const int l15 = lane & 15;
  const int hi  = lane >> 4;

  // stage this block's gather-table slice into LDS (read once from global)
  for (int t = 0; t < T; ++t)
    Atab[tid][t] = tabA[(size_t)(m0 + tid) * T + t];
  __syncthreads();

  v8f zero = {};
  v8f acc[4][4];
#pragma unroll
  for (int i = 0; i < 4; i++)
#pragma unroll
    for (int j = 0; j < 4; j++) acc[i][j] = zero;

  // global->reg staging: A row = tid (8 chunks of 8 bf16), B row = tid>>1
  const int brow  = tid >> 1;
  const int bcol0 = (tid & 1) * 32;
  const bf16_t* bptr = Bt + (size_t)(n0 + brow) * (size_t)K;

  uint4 aR[8];
  uint4 bR[4];

  auto g2r = [&](int k0) {
    int ofs = Atab[tid][k0 >> cgshift];
    unsigned msk = (ofs < 0) ? 0u : ~0u;
    int base = (ofs < 0) ? 0 : ofs;
    const uint4* ap = (const uint4*)(act + base + (k0 & cgmask));
#pragma unroll
    for (int i = 0; i < 8; i++) {
      uint4 r = ap[i];
      r.x &= msk; r.y &= msk; r.z &= msk; r.w &= msk;
      aR[i] = r;
    }
    const uint4* bp = (const uint4*)(bptr + k0 + bcol0);
#pragma unroll
    for (int i = 0; i < 4; i++) bR[i] = bp[i];
  };
  auto r2l = [&](int buf) {
#pragma unroll
    for (int i = 0; i < 8; i++) *(uint4*)(&As[buf][tid][i * 8]) = aR[i];
#pragma unroll
    for (int i = 0; i < 4; i++) *(uint4*)(&Bs[buf][brow][bcol0 + i * 8]) = bR[i];
  };

  g2r(0);
  r2l(0);
  __syncthreads();

  int buf = 0;
  for (int k0 = 0; k0 < K; k0 += BK, buf ^= 1) {
    int kn = k0 + BK;
    bool more = kn < K;
    if (more) {
      g2r(kn);                                   // overlap global latency with WMMA
      __builtin_prefetch(bptr + kn + BK, 0, 1);  // global_prefetch next B tile
    }
#pragma unroll
    for (int kk = 0; kk < BK; kk += 32) {
      FragBF a[4], b[4];
      // A 16x32 bf16 frag: lanes 0-15 row M=l, K {0..7,16..23}; lanes 16-31 K {8..15,24..31}
#pragma unroll
      for (int i = 0; i < 4; i++) {
        const bf16_t* ap = &As[buf][wm + i * 16 + l15][kk + hi * 8];
        a[i].h[0] = *(const v8bf*)ap;
        a[i].h[1] = *(const v8bf*)(ap + 16);
      }
      // B 32x16 bf16 frag: lanes 0-15 col N=l, K {0..7,16..23}; lanes 16-31 K {8..15,24..31}
#pragma unroll
      for (int j = 0; j < 4; j++) {
        const bf16_t* bp = &Bs[buf][wn + j * 16 + l15][kk + hi * 8];
        b[j].h[0] = *(const v8bf*)bp;
        b[j].h[1] = *(const v8bf*)(bp + 16);
      }
#pragma unroll
      for (int i = 0; i < 4; i++)
#pragma unroll
        for (int j = 0; j < 4; j++)
          acc[i][j] = __builtin_amdgcn_wmma_f32_16x16x32_bf16(
              false, a[i].v, false, b[j].v, (short)0, acc[i][j], false, false);
    }
    if (more) r2l(buf ^ 1);
    __syncthreads();
  }

  // Epilogue. C/D layout: VGPR r -> M = r (lanes<16) / 8+r (lanes>=16), N = lane&15
#pragma unroll
  for (int i = 0; i < 4; i++) {
#pragma unroll
    for (int j = 0; j < 4; j++) {
      int gn = n0 + wn + j * 16 + l15;
      float bv = bias ? bias[gn] : 0.f;
#pragma unroll
      for (int r = 0; r < 8; r++) {
        int gm = m0 + wm + i * 16 + hi * 8 + r;
        long base = rowdst ? (long)rowdst[gm] : (long)gm * (long)N;
        float v = alpha * acc[i][j][r] + bv;
        if (skip) v += skip[(long)gm * (long)N + gn];
        if (outF) outF[base + gn] = v;
        if (outB) outB[base + gn] = (bf16_t)(reluB ? fmaxf(v, 0.f) : v);
      }
    }
  }
}

// ---------------------------------------------------------------------------
// Weight transforms -> Bt[n][k] bf16 (K-contiguous)
// ---------------------------------------------------------------------------
// enc conv1: w [256,3,4,4] OIHW -> K=64 padded (k = tap*4 + ci, ci slot 3 = 0)
__global__ void wt_conv1(bf16_t* Bt, const float* __restrict__ w) {
  int e = blockIdx.x * blockDim.x + threadIdx.x;   // 256*64
  if (e >= 256 * 64) return;
  int o = e >> 6, k = e & 63, t = k >> 2, ci = k & 3;
  int kh = t >> 2, kw = t & 3;
  float v = (ci < 3) ? w[((o * 3 + ci) * 4 + kh) * 4 + kw] : 0.f;
  Bt[e] = (bf16_t)v;
}

// generic OIHW (O=256) -> Bt[o][(kh*KW+kw)*I + i]
__global__ void wt_oihw(bf16_t* Bt, const float* __restrict__ w, int I, int KH, int KW) {
  int K = I * KH * KW;
  int e = blockIdx.x * blockDim.x + threadIdx.x;   // 256*K
  if (e >= 256 * K) return;
  int o = e / K, k = e % K, t = k / I, i = k % I;
  int kh = t / KW, kw = t % KW;
  Bt[e] = (bf16_t)w[((o * I + i) * KH + kh) * KW + kw];
}

// codes [512,256] -> bf16 copy + 0.5*|c|^2
__global__ void wt_codes(bf16_t* Bt, float* hnorm, const float* __restrict__ codes) {
  __shared__ float red[256];
  int n = blockIdx.x, c = threadIdx.x;
  float v = codes[n * 256 + c];
  Bt[n * 256 + c] = (bf16_t)v;
  red[c] = v * v;
  __syncthreads();
  for (int s = 128; s > 0; s >>= 1) { if (c < s) red[c] += red[c + s]; __syncthreads(); }
  if (c == 0) hnorm[n] = 0.5f * red[0];
}

// convT1 weights [in=256,out=256,4,4] -> per parity class Bt[o][t*256+ci], t=(ty*2+tx)
__global__ void wt_convt1(bf16_t* Bt, const float* __restrict__ w, int py, int px) {
  int e = blockIdx.x * blockDim.x + threadIdx.x;   // 256*1024
  if (e >= 256 * 1024) return;
  int o = e >> 10, k = e & 1023, t = k >> 8, ci = k & 255;
  int ty = t >> 1, tx = t & 1;
  int kh = (py == 0) ? (ty == 0 ? 1 : 3) : (ty == 0 ? 0 : 2);
  int kw = (px == 0) ? (tx == 0 ? 1 : 3) : (tx == 0 ? 0 : 2);
  Bt[e] = (bf16_t)w[ci * 4096 + o * 16 + kh * 4 + kw];
}

// ---------------------------------------------------------------------------
// Gather tables (A-row tap offsets into NHWC bf16 activations, C=256)
// ---------------------------------------------------------------------------
__global__ void build_id_tab(int* tab, int M, int stride) {
  int m = blockIdx.x * blockDim.x + threadIdx.x;
  if (m < M) tab[m] = m * stride;
}

__global__ void build_conv_tab(int* tab, int H, int W, int KH, int KW,
                               int stride, int pad, int Hin, int Win) {
  int m = blockIdx.x * blockDim.x + threadIdx.x;   // over 32*H*W
  if (m >= 32 * H * W) return;
  int x = m % W, y = (m / W) % H, n = m / (W * H);
  for (int t = 0; t < KH * KW; t++) {
    int kh = t / KW, kw = t % KW;
    int iy = y * stride - pad + kh, ix = x * stride - pad + kw;
    tab[m * KH * KW + t] =
        (iy >= 0 && iy < Hin && ix >= 0 && ix < Win) ? ((n * Hin + iy) * Win + ix) * 256 : -1;
  }
}

// convT1 class tables: m=(n,y0,x0) over 32*32*32; 4 taps; rowdst scatter into 64x64 NHWC
__global__ void build_convt_tab(int* tab, int* rowdst, int py, int px) {
  int m = blockIdx.x * blockDim.x + threadIdx.x;
  if (m >= 32 * 32 * 32) return;
  int x0 = m & 31, y0 = (m >> 5) & 31, n = m >> 10;
  rowdst[m] = ((n * 64 + 2 * y0 + py) * 64 + 2 * x0 + px) * 256;
  for (int ty = 0; ty < 2; ty++)
    for (int tx = 0; tx < 2; tx++) {
      int iy = (py == 0) ? (ty == 0 ? y0 : y0 - 1) : (ty == 0 ? y0 + 1 : y0);
      int ix = (px == 0) ? (tx == 0 ? x0 : x0 - 1) : (tx == 0 ? x0 + 1 : x0);
      tab[m * 4 + ty * 2 + tx] =
          (iy >= 0 && iy < 32 && ix >= 0 && ix < 32) ? ((n * 32 + iy) * 32 + ix) * 256 : -1;
    }
}

// ---------------------------------------------------------------------------
// conv1 im2col: x NCHW fp32 [32,3,128,128] -> A1 bf16 [131072][64], k = tap*4+ci
// ---------------------------------------------------------------------------
__global__ void im2col1(bf16_t* A1, const float* __restrict__ x) {
  int e = blockIdx.x * blockDim.x + threadIdx.x;   // 131072*64
  if (e >= 131072 * 64) return;
  int k = e & 63, m = e >> 6;
  int ox = m & 63, oy = (m >> 6) & 63, n = m >> 12;
  int t = k >> 2, ci = k & 3;
  int kh = t >> 2, kw = t & 3;
  int iy = 2 * oy + kh - 1, ix = 2 * ox + kw - 1;
  float v = (ci < 3 && iy >= 0 && iy < 128 && ix >= 0 && ix < 128)
                ? x[((n * 3 + ci) * 128 + iy) * 128 + ix] : 0.f;
  A1[e] = (bf16_t)v;
}

// ---------------------------------------------------------------------------
// VQ argmin over 512 scores per row (one wave per row), ties -> lowest index
// ---------------------------------------------------------------------------
__global__ void argmin512(const float* __restrict__ scores, int* __restrict__ idx) {
  int lane = threadIdx.x & 31;
  int row = (blockIdx.x * blockDim.x + threadIdx.x) >> 5;
  const float* s = scores + (long)row * 512;
  float best = 3.4e38f; int bi = 0;
  for (int j = lane; j < 512; j += 32) { float v = s[j]; if (v < best) { best = v; bi = j; } }
  for (int d = 16; d > 0; d >>= 1) {
    float ov = __shfl_xor(best, d, 32);
    int   oi = __shfl_xor(bi,   d, 32);
    if (ov < best || (ov == best && oi < bi)) { best = ov; bi = oi; }
  }
  if (lane == 0) idx[row] = bi;
}

// zdec gather: zd f32 NHWC + relu'd bf16 copy (decoder resblock input)
__global__ void gather_codes(float* zd, bf16_t* zdrb, const float* __restrict__ codes,
                             const int* __restrict__ idx) {
  int e = blockIdx.x * blockDim.x + threadIdx.x;   // 32768*256
  if (e >= 32768 * 256) return;
  int m = e >> 8, c = e & 255;
  float v = codes[idx[m] * 256 + c];
  zd[e] = v;
  zdrb[e] = (bf16_t)fmaxf(v, 0.f);
}

// NHWC f32 -> NCHW f32 (N=32)
__global__ void nhwc_to_nchw(float* __restrict__ dst, const float* __restrict__ src,
                             int C, int H, int W) {
  int e = blockIdx.x * blockDim.x + threadIdx.x;   // 32*C*H*W
  if (e >= 32 * C * H * W) return;
  int x = e % W, y = (e / W) % H, c = (e / (W * H)) % C, n = e / (W * H * C);
  dst[e] = src[((n * H + y) * W + x) * C + c];
}

// ---------------------------------------------------------------------------
// convT2: 256->3, k4 s2 p1 (one wave per output pixel, VALU; ~1% of FLOPs)
// u1: NHWC f32 [32,64,64,256]; w: [256,3,4,4]; xhat: NCHW [32,3,128,128]
// ---------------------------------------------------------------------------
__global__ void convt2(float* __restrict__ xhat, const float* __restrict__ u1,
                       const float* __restrict__ w, const float* __restrict__ b) {
  int lane = threadIdx.x & 31;
  int pix = (blockIdx.x * blockDim.x + threadIdx.x) >> 5;   // 32*128*128
  if (pix >= 32 * 128 * 128) return;
  int ox = pix & 127, oy = (pix >> 7) & 127, n = pix >> 14;
  float a0 = 0.f, a1 = 0.f, a2 = 0.f;
  for (int ty = 0; ty < 2; ty++)
    for (int tx = 0; tx < 2; tx++) {
      int kh = ((oy & 1) == 0) ? (ty == 0 ? 1 : 3) : (ty == 0 ? 0 : 2);
      int kw = ((ox & 1) == 0) ? (tx == 0 ? 1 : 3) : (tx == 0 ? 0 : 2);
      int iy = (oy + 1 - kh) >> 1, ix = (ox + 1 - kw) >> 1;
      if (iy < 0 || iy >= 64 || ix < 0 || ix >= 64) continue;
      const float* up = u1 + ((long)(n * 64 + iy) * 64 + ix) * 256;
      for (int c = lane; c < 256; c += 32) {
        float uv = up[c];
        const float* wp = w + c * 48 + kh * 4 + kw;
        a0 += uv * wp[0]; a1 += uv * wp[16]; a2 += uv * wp[32];
      }
    }
  for (int d = 16; d > 0; d >>= 1) {
    a0 += __shfl_xor(a0, d, 32); a1 += __shfl_xor(a1, d, 32); a2 += __shfl_xor(a2, d, 32);
  }
  if (lane == 0) {
    long base = (long)n * 3 * 16384 + (long)oy * 128 + ox;
    xhat[base]             = a0 + b[0];
    xhat[base + 16384]     = a1 + b[1];
    xhat[base + 2 * 16384] = a2 + b[2];
  }
}

// ---------------------------------------------------------------------------
// Host orchestration
// ---------------------------------------------------------------------------
extern "C" void kernel_launch(void* const* d_in, const int* in_sizes, int n_in,
                              void* d_out, int out_size, void* d_ws, size_t ws_size,
                              hipStream_t stream) {
  (void)in_sizes; (void)n_in; (void)out_size; (void)ws_size;
  const float* x     = (const float*)d_in[0];
  const float* codes = (const float*)d_in[1];
  const float* ew1 = (const float*)d_in[2];
  const float* eb1 = (const float*)d_in[3];
  const float* ew2 = (const float*)d_in[4];
  const float* eb2 = (const float*)d_in[5];
  const float *rw[8], *rb[8];   // blocks: enc.r1, enc.r2, dec.r1, dec.r2 ; each {w1,b1,w2,b2}
  for (int i = 0; i < 4; i++) {
    rw[2 * i + 0] = (const float*)d_in[6 + 4 * i + 0];
    rb[2 * i + 0] = (const float*)d_in[6 + 4 * i + 1];
    rw[2 * i + 1] = (const float*)d_in[6 + 4 * i + 2];
    rb[2 * i + 1] = (const float*)d_in[6 + 4 * i + 3];
  }
  const float* wt1 = (const float*)d_in[22];
  const float* bt1 = (const float*)d_in[23];
  const float* wt2 = (const float*)d_in[24];
  const float* bt2 = (const float*)d_in[25];

  float* out_xhat = (float*)d_out;
  float* out_zenc = out_xhat + (long)32 * 3 * 128 * 128;
  float* out_zdec = out_zenc + (long)32 * 256 * 32 * 32;

  char* wsp = (char*)d_ws;
  size_t off = 0;
  auto alloc = [&](size_t n) -> void* {
    void* p = wsp + off;
    off += (n + 255) & ~(size_t)255;
    return p;
  };

  bf16_t* Bt1 = (bf16_t*)alloc((size_t)256 * 64 * 2);
  bf16_t* Bt2 = (bf16_t*)alloc((size_t)256 * 4096 * 2);
  bf16_t* Br[8];
  for (int i = 0; i < 4; i++) {
    Br[2 * i + 0] = (bf16_t*)alloc((size_t)256 * 2304 * 2);
    Br[2 * i + 1] = (bf16_t*)alloc((size_t)256 * 256 * 2);
  }
  bf16_t* BtC   = (bf16_t*)alloc((size_t)512 * 256 * 2);
  float*  hnorm = (float*)alloc(512 * 4);
  bf16_t* BtT[4];
  for (int c = 0; c < 4; c++) BtT[c] = (bf16_t*)alloc((size_t)256 * 1024 * 2);

  bf16_t* A1    = (bf16_t*)alloc((size_t)131072 * 64 * 2);
  bf16_t* h1bf  = (bf16_t*)alloc((size_t)131072 * 256 * 2);
  float*  a2    = (float*) alloc((size_t)32768 * 256 * 4);
  bf16_t* a2rb  = (bf16_t*)alloc((size_t)32768 * 256 * 2);
  float*  a3    = (float*) alloc((size_t)32768 * 256 * 4);
  bf16_t* a3rb  = (bf16_t*)alloc((size_t)32768 * 256 * 2);
  bf16_t* trb   = (bf16_t*)alloc((size_t)32768 * 256 * 2);
  float*  zenc  = (float*) alloc((size_t)32768 * 256 * 4);
  bf16_t* zbf   = (bf16_t*)alloc((size_t)32768 * 256 * 2);
  float*  scores= (float*) alloc((size_t)32768 * 512 * 4);
  int*    vqidx = (int*)   alloc((size_t)32768 * 4);
  float*  zd    = (float*) alloc((size_t)32768 * 256 * 4);
  bf16_t* zdrb  = (bf16_t*)alloc((size_t)32768 * 256 * 2);
  float*  dd1   = (float*) alloc((size_t)32768 * 256 * 4);
  bf16_t* d1rb  = (bf16_t*)alloc((size_t)32768 * 256 * 2);
  bf16_t* d2bf  = (bf16_t*)alloc((size_t)32768 * 256 * 2);
  float*  u1    = (float*) alloc((size_t)131072 * 256 * 4);

  int* tab_id1   = (int*)alloc((size_t)131072 * 4);
  int* tab_id256 = (int*)alloc((size_t)32768 * 4);
  int* tab_c2    = (int*)alloc((size_t)32768 * 16 * 4);
  int* tab_r3    = (int*)alloc((size_t)32768 * 9 * 4);
  int* tabT[4];  int* rdT[4];
  for (int c = 0; c < 4; c++) {
    tabT[c] = (int*)alloc((size_t)32768 * 4 * 4);
    rdT[c]  = (int*)alloc((size_t)32768 * 4);
  }

  auto gemm = [&](const bf16_t* act, const int* tab, int T, int cgshift, const bf16_t* Bt,
                  int M, int N, int K, const float* bias, const float* skip,
                  const int* rowdst, float alpha, float* outF, bf16_t* outB, int reluB) {
    dim3 g(N / BN, M / BM);
    gemm_bf16<<<g, dim3(256), 0, stream>>>(act, tab, T, cgshift, Bt, M, N, K,
                                           bias, skip, rowdst, alpha, outF, outB, reluB);
  };

  // ---- weight / code transforms ----
  wt_conv1<<<64, 256, 0, stream>>>(Bt1, ew1);
  wt_oihw<<<4096, 256, 0, stream>>>(Bt2, ew2, 256, 4, 4);
  for (int i = 0; i < 4; i++) {
    wt_oihw<<<2304, 256, 0, stream>>>(Br[2 * i + 0], rw[2 * i + 0], 256, 3, 3);
    wt_oihw<<<256, 256, 0, stream>>>(Br[2 * i + 1], rw[2 * i + 1], 256, 1, 1);
  }
  wt_codes<<<512, 256, 0, stream>>>(BtC, hnorm, codes);
  for (int c = 0; c < 4; c++)
    wt_convt1<<<1024, 256, 0, stream>>>(BtT[c], wt1, c >> 1, c & 1);

  // ---- gather tables ----
  build_id_tab<<<512, 256, 0, stream>>>(tab_id1, 131072, 64);
  build_id_tab<<<128, 256, 0, stream>>>(tab_id256, 32768, 256);
  build_conv_tab<<<128, 256, 0, stream>>>(tab_c2, 32, 32, 4, 4, 2, 1, 64, 64);
  build_conv_tab<<<128, 256, 0, stream>>>(tab_r3, 32, 32, 3, 3, 1, 1, 32, 32);
  for (int c = 0; c < 4; c++)
    build_convt_tab<<<128, 256, 0, stream>>>(tabT[c], rdT[c], c >> 1, c & 1);

  // ---- encoder ----
  im2col1<<<32768, 256, 0, stream>>>(A1, x);
  gemm(A1,   tab_id1,   1, 6, Bt1, 131072, 256,   64, eb1, nullptr, nullptr, 1.f, nullptr, h1bf, 0);
  gemm(h1bf, tab_c2,   16, 8, Bt2,  32768, 256, 4096, eb2, nullptr, nullptr, 1.f, a2, a2rb, 1);
  gemm(a2rb, tab_r3,    9, 8, Br[0], 32768, 256, 2304, rb[0], nullptr, nullptr, 1.f, nullptr, trb, 1);
  gemm(trb,  tab_id256, 1, 8, Br[1], 32768, 256,  256, rb[1], a2, nullptr, 1.f, a3, a3rb, 1);
  gemm(a3rb, tab_r3,    9, 8, Br[2], 32768, 256, 2304, rb[2], nullptr, nullptr, 1.f, nullptr, trb, 1);
  gemm(trb,  tab_id256, 1, 8, Br[3], 32768, 256,  256, rb[3], a3, nullptr, 1.f, zenc, zbf, 0);

  // ---- vector quantization: scores = 0.5|c|^2 - z.c, argmin, gather ----
  gemm(zbf, tab_id256, 1, 8, BtC, 32768, 512, 256, hnorm, nullptr, nullptr, -1.f, scores, nullptr, 0);
  argmin512<<<4096, 256, 0, stream>>>(scores, vqidx);
  gather_codes<<<32768, 256, 0, stream>>>(zd, zdrb, codes, vqidx);
  nhwc_to_nchw<<<32768, 256, 0, stream>>>(out_zenc, zenc, 256, 32, 32);
  nhwc_to_nchw<<<32768, 256, 0, stream>>>(out_zdec, zd, 256, 32, 32);

  // ---- decoder (forward value of z_st is exactly zdec) ----
  gemm(zdrb, tab_r3,    9, 8, Br[4], 32768, 256, 2304, rb[4], nullptr, nullptr, 1.f, nullptr, trb, 1);
  gemm(trb,  tab_id256, 1, 8, Br[5], 32768, 256,  256, rb[5], zd, nullptr, 1.f, dd1, d1rb, 1);
  gemm(d1rb, tab_r3,    9, 8, Br[6], 32768, 256, 2304, rb[6], nullptr, nullptr, 1.f, nullptr, trb, 1);
  gemm(trb,  tab_id256, 1, 8, Br[7], 32768, 256,  256, rb[7], dd1, nullptr, 1.f, nullptr, d2bf, 0);

  // convT1 as 4 output-parity-class GEMMs with rowdst scatter into u1 (NHWC 64x64)
  for (int c = 0; c < 4; c++)
    gemm(d2bf, tabT[c], 4, 8, BtT[c], 32768, 256, 1024, bt1, nullptr, rdT[c], 1.f, u1, nullptr, 0);

  // convT2 -> xhat (NCHW) directly into d_out
  convt2<<<65536, 256, 0, stream>>>(out_xhat, u1, wt2, bt2);
}
